// NeuralNDCGLoss_72730976190520
// MI455X (gfx1250) — matrix operand order repaired and checked by reading, not compiled, not run on hardware
//
#include <hip/hip_runtime.h>
#include <math.h>

#define B_DIM 512
#define N_DIM 128
#define D_DIM 768
#define PITCHH 130   // f16 row pitch: dword stride 65 == 1 mod 64 banks

typedef __attribute__((ext_vector_type(16))) _Float16 v16h;
typedef __attribute__((ext_vector_type(8)))  float    v8f;

__device__ __forceinline__ float fastrcp(float x) { return __builtin_amdgcn_rcpf(x); }

// acc = M(128x128, f16 LDS, pitch PITCHH) @ x(128, f16 LDS), f32 accumulate.
// Broadcast-B trick: every column of C equals M@x. Wave w covers rows
// 16w..16w+15; lanes with (lane&15)==0 hold the result column (8 rows each).
__device__ __forceinline__ v8f matvec_core(const _Float16* __restrict__ M,
                                           const _Float16* __restrict__ x,
                                           int w, int lane) {
  v8f acc8 = {0.f, 0.f, 0.f, 0.f, 0.f, 0.f, 0.f, 0.f};
  const int m  = 16 * w + (lane & 15);
  const int hi = lane >> 4;
#pragma unroll
  for (int k0 = 0; k0 < N_DIM; k0 += 32) {
    v16h Av, Bv;
#pragma unroll
    for (int e = 0; e < 8; ++e) {
      int ka = ((e < 4) ? (2 * e) : (16 + 2 * (e - 4))) + 8 * hi;  // A 16x32 f16 layout
      Av[2 * e]     = M[m * PITCHH + k0 + ka];
      Av[2 * e + 1] = M[m * PITCHH + k0 + ka + 1];
      int kb = ((e < 4) ? (2 * e) : (16 + 2 * (e - 4))) + 8 * hi;  // B 32x16 f16 layout
      Bv[2 * e]     = x[k0 + kb];
      Bv[2 * e + 1] = x[k0 + kb + 1];
    }
    acc8 = __builtin_amdgcn_wmma_f32_16x16x32_f16(
        false, Av, false, Bv, (short)0, acc8, false, false);
  }
  return acc8;
}

// ---------------------------------------------------------------------------
// One workgroup (256 threads = 8 wave32) per batch element. Fully fused:
// scores -> neural sort -> factorized Sinkhorn (WMMA matvecs) -> ndcg.
// ---------------------------------------------------------------------------
__global__ __launch_bounds__(256) void ndcg_main(
    const float* __restrict__ q, const float* __restrict__ r,
    const float* __restrict__ labels, float* __restrict__ accum) {

  __shared__ _Float16 Ph [N_DIM * PITCHH];   // m0 (post-softmax, post-fixup), f16
  __shared__ _Float16 PhT[N_DIM * PITCHH];   // m0^T, f16
  __shared__ _Float16 uh[N_DIM], vh[N_DIM];  // f16 copies of scaling vectors
  __shared__ float s[N_DIM], yv[N_DIM], g[N_DIM], msk[N_DIM], Bm[N_DIM];
  __shared__ float u[N_DIM], v[N_DIM], tvec[N_DIM], ysort[N_DIM];
  __shared__ float red[256];
  __shared__ float cosPart, dsumS;
  __shared__ int nvS;

  const int b    = blockIdx.x;
  const int tid  = threadIdx.x;
  const int lane = tid & 31;
  const int w    = tid >> 5;

  // ---- 1. cosine similarity scores: wave per row, float4 streaming --------
  const float* qb = q + (size_t)b * N_DIM * D_DIM;
  const float* rb = r + (size_t)b * N_DIM * D_DIM;
  for (int n = w; n < N_DIM; n += 8) {
    if (n + 8 < N_DIM) {               // pull next row's cachelines in flight
      __builtin_prefetch(qb + (n + 8) * D_DIM + lane * 32, 0, 3);
      __builtin_prefetch(rb + (n + 8) * D_DIM + lane * 32, 0, 3);
    }
    const float4* q4 = (const float4*)(qb + n * D_DIM);
    const float4* r4 = (const float4*)(rb + n * D_DIM);
    float qq = 0.f, rr = 0.f, qr = 0.f;
    for (int i = lane; i < D_DIM / 4; i += 32) {
      float4 a = q4[i]; float4 c = r4[i];
      qq += a.x*a.x + a.y*a.y + a.z*a.z + a.w*a.w;
      rr += c.x*c.x + c.y*c.y + c.z*c.z + c.w*c.w;
      qr += a.x*c.x + a.y*c.y + a.z*c.z + a.w*c.w;
    }
    for (int off = 16; off > 0; off >>= 1) {
      qq += __shfl_xor(qq, off, 32);
      rr += __shfl_xor(rr, off, 32);
      qr += __shfl_xor(qr, off, 32);
    }
    if (lane == 0) {
      float den = fmaxf(sqrtf(qq), 1e-8f) * fmaxf(sqrtf(rr), 1e-8f);
      s[n] = qr * fastrcp(den);
    }
  }
  if (tid < N_DIM) {
    float yy = labels[(size_t)b * N_DIM + tid];
    yv[tid] = yy;
    float m = (yy == -1.0f) ? 1.f : 0.f;
    msk[tid] = m;
    g[tid] = (m != 0.f) ? 0.f : (exp2f(yy) - 1.f);
  }
  __syncthreads();

  // ---- cosine MSE partial + valid count -----------------------------------
  {
    float e2 = 0.f, mc = 0.f;
    if (tid < N_DIM) {
      float d = s[tid] - yv[tid];
      e2 = d * d;
      mc = msk[tid];
    }
    red[tid] = e2;
    __syncthreads();
    for (int st = 128; st > 0; st >>= 1) {
      if (tid < st) red[tid] += red[tid + st];
      __syncthreads();
    }
    if (tid == 0) cosPart = red[0];
    __syncthreads();
    red[tid] = mc;
    __syncthreads();
    for (int st = 128; st > 0; st >>= 1) {
      if (tid < st) red[tid] += red[tid + st];
      __syncthreads();
    }
    if (tid == 0) nvS = N_DIM - (int)(red[0] + 0.5f);
    __syncthreads();
  }
  const int nv = nvS;

  // ---- 2. NeuralSort: Bmat_j = sum_j' |s_j - s_j'| (mask-aware) -----------
  if (tid < N_DIM) {
    float a = 0.f;
    if (msk[tid] == 0.f) {
      float sj = s[tid];
      for (int j2 = 0; j2 < N_DIM; ++j2)
        if (msk[j2] == 0.f) a += fabsf(sj - s[j2]);
    }
    Bm[tid] = a;
  }
  __syncthreads();

  // ---- P_max row i + softmax (tau=1) -> f16 m0 and m0^T -------------------
  for (int i = w; i < N_DIM; i += 8) {
    float scal = (i < nv) ? (float)(nv - 1 - 2 * i) : 0.f;
    bool mi = msk[i] != 0.f;
    float v0[4];
    float mx = -3.4e38f;
#pragma unroll
    for (int t = 0; t < 4; ++t) {
      int j = lane + 32 * t;
      bool mj = msk[j] != 0.f;
      float pv;
      if (mi && mj)       pv = 1.0f;
      else if (mi || mj)  pv = -3.4e38f;
      else                pv = s[j] * scal - Bm[j];
      v0[t] = pv;
      mx = fmaxf(mx, pv);
    }
    for (int off = 16; off > 0; off >>= 1) mx = fmaxf(mx, __shfl_xor(mx, off, 32));
    float sm = 0.f, e[4];
#pragma unroll
    for (int t = 0; t < 4; ++t) { e[t] = __expf(v0[t] - mx); sm += e[t]; }
    for (int off = 16; off > 0; off >>= 1) sm += __shfl_xor(sm, off, 32);
    float inv = fastrcp(sm);
#pragma unroll
    for (int t = 0; t < 4; ++t) {
      int j = lane + 32 * t;
      _Float16 val = (_Float16)(e[t] * inv);
      Ph [i * PITCHH + j] = val;
      PhT[j * PITCHH + i] = val;
    }
  }

  const int j0 = tid & 127;
  const int hh = tid >> 7;
  // init scaling vectors
  if (tid < N_DIM) { u[tid] = 1.f; v[tid] = 1.f; uh[tid] = (_Float16)1.f; }
  __syncthreads();

  // ---- pre-sinkhorn mask fixup on m0 (identity when unpadded) -------------
  {
    bool mj = msk[j0] != 0.f;
    for (int rr2 = hh * 64; rr2 < hh * 64 + 64; ++rr2) {
      bool mi = msk[rr2] != 0.f;
      if (mi || mj) {
        _Float16 val = (_Float16)((mi && mj) ? 1.f : 0.f);
        Ph [rr2 * PITCHH + j0] = val;
        PhT[j0 * PITCHH + rr2] = val;
      }
    }
  }
  __syncthreads();

  // ---- 3. Factorized Sinkhorn: m = diag(u) m0 diag(v); 50 iterations ------
  // colsum_j = v_j * (m0^T u)_j ; rowsum_i = u_i * (m0 v)_i.
  // Scaling update fused into the WMMA epilogue (result-holding lanes own
  // disjoint rows) -> 2 barriers per iteration.
  for (int it = 0; it < 50; ++it) {
    {
      v8f t8 = matvec_core(PhT, uh, w, lane);      // t = m0^T u
      if ((lane & 15) == 0) {
        int baseRow = 16 * w + 8 * (lane >> 4);
#pragma unroll
        for (int e = 0; e < 8; ++e) {
          int rowi = baseRow + e;
          float old = v[rowi];
          float nvv = old * fastrcp(fmaxf(old * t8[e], 1e-10f));
          v[rowi] = nvv; vh[rowi] = (_Float16)nvv;
        }
      }
      __syncthreads();
    }
    {
      v8f t8 = matvec_core(Ph, vh, w, lane);       // s = m0 v'
      if ((lane & 15) == 0) {
        int baseRow = 16 * w + 8 * (lane >> 4);
#pragma unroll
        for (int e = 0; e < 8; ++e) {
          int rowi = baseRow + e;
          float old = u[rowi];
          float nuu = old * fastrcp(fmaxf(old * t8[e], 1e-10f));
          u[rowi] = nuu; uh[rowi] = (_Float16)nuu;
        }
      }
      __syncthreads();
    }
  }

  // ---- 4. gt = (1-mask) * u ⊙ (m0 @ (v ⊙ g)) ------------------------------
  if (tid < N_DIM) vh[tid] = (_Float16)(v[tid] * g[tid]);
  __syncthreads();
  {
    v8f t8 = matvec_core(Ph, vh, w, lane);
    if ((lane & 15) == 0) {
      int baseRow = 16 * w + 8 * (lane >> 4);
#pragma unroll
      for (int e = 0; e < 8; ++e) tvec[baseRow + e] = t8[e];
    }
  }
  __syncthreads();

  // ---- discounted DCG sum -------------------------------------------------
  {
    float dv = 0.f;
    if (tid < N_DIM) {
      float gti = (msk[tid] != 0.f) ? 0.f : u[tid] * tvec[tid];
      dv = gti * fastrcp(log2f((float)tid + 2.0f));
    }
    red[tid] = dv;
    __syncthreads();
    for (int st = 128; st > 0; st >>= 1) {
      if (tid < st) red[tid] += red[tid + st];
      __syncthreads();
    }
    if (tid == 0) dsumS = red[0];
    __syncthreads();
  }

  // ---- IDCG: bitonic sort y descending, then powered-gain DCG -------------
  if (tid < N_DIM) ysort[tid] = yv[tid];
  __syncthreads();
  for (int size = 2; size <= N_DIM; size <<= 1) {
    for (int stride = size >> 1; stride > 0; stride >>= 1) {
      if (tid < N_DIM) {
        int p = tid ^ stride;
        if (p > tid) {
          float a = ysort[tid], c = ysort[p];
          bool desc = ((tid & size) == 0);
          bool sw = desc ? (a < c) : (a > c);
          if (sw) { ysort[tid] = c; ysort[p] = a; }
        }
      }
      __syncthreads();
    }
  }
  {
    float iv2 = 0.f;
    if (tid < N_DIM)
      iv2 = (exp2f(ysort[tid]) - 1.0f) * fastrcp(log2f((float)tid + 2.0f));
    red[tid] = iv2;
    __syncthreads();
    for (int st = 128; st > 0; st >>= 1) {
      if (tid < st) red[tid] += red[tid + st];
      __syncthreads();
    }
    if (tid == 0) {
      float idcg = red[0];
      float ndcg = (idcg == 0.0f) ? 0.0f : (dsumS / (idcg + 1e-10f));
      atomicAdd(&accum[0], ndcg);
      if (idcg != 0.0f) atomicAdd(&accum[1], 1.0f);
      atomicAdd(&accum[2], cosPart);
    }
  }
}

__global__ void ndcg_init(float* accum) {
  if (threadIdx.x < 3) accum[threadIdx.x] = 0.f;
}

__global__ void ndcg_finalize(const float* __restrict__ accum, float* __restrict__ out) {
  float meanN = accum[0] / fmaxf(accum[1], 1.0f);
  float cosL  = accum[2] / (float)(B_DIM * N_DIM);
  out[0] = 0.9f * (1.0f - meanN) + 0.1f * cosL;
}

extern "C" void kernel_launch(void* const* d_in, const int* in_sizes, int n_in,
                              void* d_out, int out_size, void* d_ws, size_t ws_size,
                              hipStream_t stream) {
  const float* q = (const float*)d_in[0];
  const float* r = (const float*)d_in[1];
  const float* y = (const float*)d_in[2];
  float* accum = (float*)d_ws;
  float* out   = (float*)d_out;
  (void)in_sizes; (void)n_in; (void)out_size; (void)ws_size;

  ndcg_init<<<1, 32, 0, stream>>>(accum);
  ndcg_main<<<B_DIM, 256, 0, stream>>>(q, r, y, accum);
  ndcg_finalize<<<1, 1, 0, stream>>>(accum, out);
}
